// MHA_13950053777893
// MI455X (gfx1250) — compile-verified, matrix operand order checked by reading, not run on hardware
//
#include <hip/hip_runtime.h>

// ---------------------------------------------------------------------------
// GQA flash-attention forward for gfx1250 (MI455X), wave32 + WMMA bf16 + TDM.
//   q  : (B, SQ, H, D)  f32      kv : (B, SK, 2, HKV, D) f32
//   key_padding_mask : (B, SK) bool (1 byte)     out: (B, SQ, H, D) f32
// Matrix path is compute-bound (~820 FLOP/byte) so f32 inputs are cast to
// bf16 (K=32 per WMMA vs K=4 for the f32 WMMA). K tiles are staged by the
// Tensor Data Mover (async, TENSORcnt) with descriptor-driven LDS padding;
// V is staged global->reg->LDS (transposed, packed b32 stores) so its
// latency hides under WMMAs. Double buffered: one barrier per 32-key tile.
// Score WMMAs run as 4-deep accumulator chains (no D->A/B hazard) so LDS
// fragment loads clause up instead of wait-per-WMMA.
// ---------------------------------------------------------------------------

typedef __attribute__((ext_vector_type(16))) __bf16 v16bf;
typedef __attribute__((ext_vector_type(8)))  __bf16 v8bf;
typedef __attribute__((ext_vector_type(8)))  float  v8f;
typedef __attribute__((ext_vector_type(4)))  unsigned int u32x4;
typedef __attribute__((ext_vector_type(8)))  int i32x8;
typedef __attribute__((ext_vector_type(4)))  int i32x4;

#ifndef __has_builtin
#define __has_builtin(x) 0
#endif
#if __has_builtin(__builtin_amdgcn_tensor_load_to_lds) && \
    __has_builtin(__builtin_amdgcn_s_wait_tensorcnt)
#define HAVE_TDM 1
#else
#define HAVE_TDM 0
#endif

#define B_    2
#define SQ_   2048
#define SK_   2048
#define H_    16
#define HKV_  4
#define D_    128
#define NEG_  (-10000.0f)
#define SCALE_ 0.08838834764831845f   // 1/sqrt(128)

#define BQ 128   // q rows per block (8 waves x 16)
#define BK 32    // key tile
#define NT_ (SK_ / BK)

// ------------------------------ prep kernels -------------------------------

__global__ void conv_q_kernel(const float* __restrict__ q,
                              __bf16* __restrict__ qbf, int n) {
  int i = blockIdx.x * blockDim.x + threadIdx.x;
  if (i < n) qbf[i] = (__bf16)q[i];
}

// de-interleave kv (B,SK,2,HKV,D) -> kbf/vbf laid out (B,HKV,SK,D)
__global__ void conv_kv_kernel(const float* __restrict__ kv,
                               __bf16* __restrict__ kbf,
                               __bf16* __restrict__ vbf, int n) {
  int i = blockIdx.x * blockDim.x + threadIdx.x;
  if (i >= n) return;
  int d  = i % D_;
  int s  = (i / D_) % SK_;
  int hk = (i / (D_ * SK_)) % HKV_;
  int b  = i / (D_ * SK_ * HKV_);
  long long src = ((((long long)b * SK_ + s) * 2 + 0) * HKV_ + hk) * D_ + d;
  kbf[i] = (__bf16)kv[src];
  vbf[i] = (__bf16)kv[src + (long long)HKV_ * D_];
}

__global__ void seqlen_kernel(const unsigned char* __restrict__ mask,
                              int* __restrict__ lens) {
  __shared__ int red[256];
  int b = blockIdx.x;
  int acc = 0;
  for (int j = threadIdx.x; j < SK_; j += 256) acc += mask[b * SK_ + j] ? 1 : 0;
  red[threadIdx.x] = acc;
  __syncthreads();
  for (int off = 128; off > 0; off >>= 1) {
    if ((int)threadIdx.x < off) red[threadIdx.x] += red[threadIdx.x + off];
    __syncthreads();
  }
  if (threadIdx.x == 0) lens[b] = red[0];
}

// ------------------------------ attention ----------------------------------

__device__ __forceinline__ v16bf mk16(v8bf lo, v8bf hi) {
  v16bf r;
#pragma unroll
  for (int i = 0; i < 8; ++i) { r[i] = lo[i]; r[i + 8] = hi[i]; }
  return r;
}

__device__ __forceinline__ unsigned int pack2(__bf16 lo, __bf16 hi) {
  unsigned int l = __builtin_bit_cast(unsigned short, lo);
  unsigned int h = __builtin_bit_cast(unsigned short, hi);
  return l | (h << 16);
}

#if HAVE_TDM
// TDM descriptor: load a BK x D_ bf16 tile (row = D_*2 = 256B) from a
// (SK_ x D_) bf16 tensor into LDS, inserting 16B of padding after every
// 256B so rows land at a 272B stride (= Kl[.][136] layout, conflict-free).
__device__ __forceinline__ void tdm_load_k_tile(const __bf16* gsrc,
                                                void* ldst) {
  unsigned long long ga = (unsigned long long)(size_t)gsrc;
  unsigned int lds = (unsigned int)(size_t)ldst;  // addr[31:0] = LDS offset
  u32x4 g0;
  g0[0] = 1u;                                   // count=1, user-mode D#
  g0[1] = lds;                                  // lds_addr
  g0[2] = (unsigned int)ga;                     // global_addr[31:0]
  g0[3] = (unsigned int)((ga >> 32) & 0x01ffffffu) | (2u << 30);  // type=2
  i32x8 g1;
  g1[0] = (1 << 16)       // data_size = 2 bytes (bf16)
        | (1 << 20)       // pad_enable
        | (5 << 22)       // pad_interval: every 64 DWORDs (256B)
        | (3 << 25);      // pad_amount : 4 DWORDs (16B)
  g1[1] = (int)(((unsigned)D_) << 16);   // tensor_dim0 = 128 elements
  g1[2] = (int)(((unsigned)SK_) << 16);  // tensor_dim1 = 2048 rows
  g1[3] = (int)(((unsigned)D_) << 16);   // tile_dim0   = 128 elements
  g1[4] = BK;                            // tile_dim1   = 32 rows
  g1[5] = D_;                            // tensor_dim0_stride = 128
  g1[6] = 0;
  g1[7] = 0;
  i32x4 z4 = {0, 0, 0, 0};
#if __clang_major__ >= 23
  i32x8 z8 = {0, 0, 0, 0, 0, 0, 0, 0};
  __builtin_amdgcn_tensor_load_to_lds(g0, g1, z4, z4, z8, 0);
#else
  __builtin_amdgcn_tensor_load_to_lds(g0, g1, z4, z4, 0);
#endif
}
#endif

__global__ __launch_bounds__(256, 1) void fa_fwd_kernel(
    const __bf16* __restrict__ qbf, const __bf16* __restrict__ kbf,
    const __bf16* __restrict__ vbf, const unsigned char* __restrict__ mask,
    const int* __restrict__ lens, float* __restrict__ out) {
  // double-buffered K: rows padded 128->136 halfs (272B, 16B aligned)
  __shared__ __align__(16) __bf16 Kl[2][BK][136];
  // double-buffered V, transposed: Vt[d][s], stride 40 halfs = 80B
  __shared__ __align__(16) __bf16 Vt[2][D_][40];
  // per-wave probability tile, 16 x 32 (stride 40)
  __shared__ __align__(16) __bf16 Pl[8][16][40];

  const int tid   = threadIdx.x;
  const int wave  = tid >> 5;
  const int lane  = tid & 31;
  const int lmod  = lane & 15;
  const int lhi   = lane >> 4;
  const int kbase = lhi * 8;

  const int h  = blockIdx.y;
  const int b  = blockIdx.z;
  const int hk = h / (H_ / HKV_);
  const int qrow0 = blockIdx.x * BQ + wave * 16;
  const int sk = lens[b];

  // --- Q fragments (A-layout) held in registers for the whole kernel
  const __bf16* qrow =
      qbf + (((long long)b * SQ_ + (qrow0 + lmod)) * H_ + h) * D_;
  v16bf qa[4];
#pragma unroll
  for (int f = 0; f < 4; ++f) {
    const __bf16* p = qrow + f * 32 + kbase;
    qa[f] = mk16(*(const v8bf*)p, *(const v8bf*)(p + 16));
  }

  v8f o[8];
#pragma unroll
  for (int dt = 0; dt < 8; ++dt) o[dt] = {};
  float mrow[8], lrow[8];
#pragma unroll
  for (int r = 0; r < 8; ++r) { mrow[r] = -1.0e30f; lrow[r] = 0.0f; }

  const __bf16* Kg = kbf + ((long long)(b * HKV_ + hk) * SK_) * D_;
  const __bf16* Vg = vbf + ((long long)(b * HKV_ + hk) * SK_) * D_;
  const unsigned char* mrw = mask + b * SK_;

  // V staging partition: each thread stages key rows {sr2, sr2+1} x 8 d-vals
  const int sr2 = (tid >> 4) * 2;   // even key row 0..30
  const int sc8 = (tid & 15) * 8;   // d offset 0..120

  // ---------------- prologue: stage tile 0 into buffer 0 ----------------
#if HAVE_TDM
  if (wave == 0) tdm_load_k_tile(Kg, &Kl[0][0][0]);
#else
  {
    const int srow = tid >> 3;
    const int scol = (tid & 7) * 16;
    const __bf16* ksrc = Kg + (long long)srow * D_ + scol;
    *(v8bf*)&Kl[0][srow][scol]     = *(const v8bf*)ksrc;
    *(v8bf*)&Kl[0][srow][scol + 8] = *(const v8bf*)(ksrc + 8);
  }
#endif
  {
    const __bf16* vsrc = Vg + (long long)sr2 * D_ + sc8;
    v8bf r0 = *(const v8bf*)vsrc;
    v8bf r1 = *(const v8bf*)(vsrc + D_);
#pragma unroll
    for (int j = 0; j < 8; ++j)
      *(unsigned int*)&Vt[0][sc8 + j][sr2] = pack2(r0[j], r1[j]);
  }

  // ---------------- main loop: one barrier per 32-key tile ----------------
  for (int t = 0; t < NT_; ++t) {
    const int cur = t & 1;
    const int nxt = cur ^ 1;
    const int s0 = t * BK;
    const bool more = (t + 1) < NT_;

#if HAVE_TDM
    if (wave == 0) __builtin_amdgcn_s_wait_tensorcnt(0);
#endif
    __syncthreads();  // staging of tile t complete & visible to all waves

    // kick off staging of tile t+1 (writes only the other buffer)
    v8bf vr0 = {}, vr1 = {};
    if (more) {
#if HAVE_TDM
      if (wave == 0)
        tdm_load_k_tile(Kg + (long long)(s0 + BK) * D_, &Kl[nxt][0][0]);
#else
      const int srow = tid >> 3;
      const int scol = (tid & 7) * 16;
      const __bf16* ksrc = Kg + (long long)(s0 + BK + srow) * D_ + scol;
      *(v8bf*)&Kl[nxt][srow][scol]     = *(const v8bf*)ksrc;
      *(v8bf*)&Kl[nxt][srow][scol + 8] = *(const v8bf*)(ksrc + 8);
#endif
      const __bf16* vsrc = Vg + (long long)(s0 + BK + sr2) * D_ + sc8;
      vr0 = *(const v8bf*)vsrc;        // global loads issue now,
      vr1 = *(const v8bf*)(vsrc + D_); // ds stores deferred to after compute
    }

    // --- scores: two sequential 4-deep WMMA accumulator chains.
    //     Accumulator (D->C) chaining has no hazard, so each chain needs one
    //     dscnt wait for its 8 clause-issued ds_load_b128s, then 4 WMMAs
    //     run back-to-back.
    v8f sc0, sc1;
    {
      const __bf16* kp = &Kl[cur][lmod][kbase];
      v16bf k0 = mk16(*(const v8bf*)(kp +  0), *(const v8bf*)(kp +  16));
      v16bf k1 = mk16(*(const v8bf*)(kp + 32), *(const v8bf*)(kp +  48));
      v16bf k2 = mk16(*(const v8bf*)(kp + 64), *(const v8bf*)(kp +  80));
      v16bf k3 = mk16(*(const v8bf*)(kp + 96), *(const v8bf*)(kp + 112));
      v8f a = {};
      a = __builtin_amdgcn_wmma_f32_16x16x32_bf16(false, qa[0], false, k0,
                                                  (short)0, a, false, false);
      a = __builtin_amdgcn_wmma_f32_16x16x32_bf16(false, qa[1], false, k1,
                                                  (short)0, a, false, false);
      a = __builtin_amdgcn_wmma_f32_16x16x32_bf16(false, qa[2], false, k2,
                                                  (short)0, a, false, false);
      a = __builtin_amdgcn_wmma_f32_16x16x32_bf16(false, qa[3], false, k3,
                                                  (short)0, a, false, false);
      sc0 = a;
    }
    {
      const __bf16* kp = &Kl[cur][16 + lmod][kbase];
      v16bf k0 = mk16(*(const v8bf*)(kp +  0), *(const v8bf*)(kp +  16));
      v16bf k1 = mk16(*(const v8bf*)(kp + 32), *(const v8bf*)(kp +  48));
      v16bf k2 = mk16(*(const v8bf*)(kp + 64), *(const v8bf*)(kp +  80));
      v16bf k3 = mk16(*(const v8bf*)(kp + 96), *(const v8bf*)(kp + 112));
      v8f a = {};
      a = __builtin_amdgcn_wmma_f32_16x16x32_bf16(false, qa[0], false, k0,
                                                  (short)0, a, false, false);
      a = __builtin_amdgcn_wmma_f32_16x16x32_bf16(false, qa[1], false, k1,
                                                  (short)0, a, false, false);
      a = __builtin_amdgcn_wmma_f32_16x16x32_bf16(false, qa[2], false, k2,
                                                  (short)0, a, false, false);
      a = __builtin_amdgcn_wmma_f32_16x16x32_bf16(false, qa[3], false, k3,
                                                  (short)0, a, false, false);
      sc1 = a;
    }

    // --- scale + padding NEG + causal NEG (value masking, never skip tiles:
    //     matches reference softmax semantics for fully-masked rows)
    const int col0 = s0 + lmod;
    const float pad0 = mrw[col0] ? 0.0f : NEG_;
    const float pad1 = mrw[col0 + 16] ? 0.0f : NEG_;
#pragma unroll
    for (int r = 0; r < 8; ++r) {
      const int lim = qrow0 + r + 8 * lhi + sk - SQ_;
      float a = sc0[r] * SCALE_ + pad0;
      float c = sc1[r] * SCALE_ + pad1;
      if (col0 > lim) a = NEG_;
      if (col0 + 16 > lim) c = NEG_;
      sc0[r] = a;
      sc1[r] = c;
    }

    // --- online softmax (reduce across 16-lane groups, wave32 shuffles)
#pragma unroll
    for (int r = 0; r < 8; ++r) {
      float tmax = fmaxf(sc0[r], sc1[r]);
#pragma unroll
      for (int x = 1; x < 16; x <<= 1) tmax = fmaxf(tmax, __shfl_xor(tmax, x));
      const float mn   = fmaxf(mrow[r], tmax);
      const float corr = __expf(mrow[r] - mn);
      mrow[r] = mn;
      lrow[r] *= corr;
#pragma unroll
      for (int dt = 0; dt < 8; ++dt) o[dt][r] *= corr;
      const float p0 = __expf(sc0[r] - mn);
      const float p1 = __expf(sc1[r] - mn);
      float ps = p0 + p1;
#pragma unroll
      for (int x = 1; x < 16; x <<= 1) ps += __shfl_xor(ps, x);
      lrow[r] += ps;
      Pl[wave][r + 8 * lhi][lmod]      = (__bf16)p0;
      Pl[wave][r + 8 * lhi][16 + lmod] = (__bf16)p1;
    }

    // --- O += P x V (A from wave-private LDS P, B from transposed V)
    const __bf16* pp = &Pl[wave][lmod][kbase];
    v16bf pa = mk16(*(const v8bf*)pp, *(const v8bf*)(pp + 16));
#pragma unroll
    for (int g = 0; g < 2; ++g) {
      v16bf vf[4];
#pragma unroll
      for (int i = 0; i < 4; ++i) {
        const __bf16* vp = &Vt[cur][(g * 4 + i) * 16 + lmod][kbase];
        vf[i] = mk16(*(const v8bf*)vp, *(const v8bf*)(vp + 16));
      }
#pragma unroll
      for (int i = 0; i < 4; ++i)
        o[g * 4 + i] = __builtin_amdgcn_wmma_f32_16x16x32_bf16(
            false, pa, false, vf[i], (short)0, o[g * 4 + i], false, false);
    }

    // --- finish staging V tile t+1 (packed b32 ds stores after compute)
    if (more) {
#pragma unroll
      for (int j = 0; j < 8; ++j)
        *(unsigned int*)&Vt[nxt][sc8 + j][sr2] = pack2(vr0[j], vr1[j]);
    }
  }

  // --- epilogue: normalize and store (B, SQ, H, D) f32
#pragma unroll
  for (int r = 0; r < 8; ++r) {
    const int row = qrow0 + r + 8 * lhi;
    const float inv = 1.0f / lrow[r];
    float* op = out + (((long long)b * SQ_ + row) * H_ + h) * D_;
#pragma unroll
    for (int dt = 0; dt < 8; ++dt) op[dt * 16 + lmod] = o[dt][r] * inv;
  }
}

// ------------------------------ launcher -----------------------------------

extern "C" void kernel_launch(void* const* d_in, const int* in_sizes, int n_in,
                              void* d_out, int out_size, void* d_ws,
                              size_t ws_size, hipStream_t stream) {
  const float* q  = (const float*)d_in[0];
  const float* kv = (const float*)d_in[1];
  const unsigned char* mask = (const unsigned char*)d_in[2];  // jax bool = 1B
  float* out = (float*)d_out;

  const long long nQ  = (long long)B_ * SQ_ * H_ * D_;    // 8,388,608
  const long long nKV = (long long)B_ * HKV_ * SK_ * D_;  // 2,097,152

  __bf16* qbf  = (__bf16*)d_ws;
  __bf16* kbf  = qbf + nQ;
  __bf16* vbf  = kbf + nKV;
  int*    lens = (int*)(vbf + nKV);   // ~25.2 MB total workspace

  conv_q_kernel<<<(int)((nQ + 255) / 256), 256, 0, stream>>>(q, qbf, (int)nQ);
  conv_kv_kernel<<<(int)((nKV + 255) / 256), 256, 0, stream>>>(kv, kbf, vbf,
                                                               (int)nKV);
  seqlen_kernel<<<B_, 256, 0, stream>>>(mask, lens);

  dim3 grid(SQ_ / BQ, H_, B_);
  fa_fwd_kernel<<<grid, 256, 0, stream>>>(qbf, kbf, vbf, mask, lens, out);
}
